// LayoutGraphModel_87428354277525
// MI455X (gfx1250) — matrix operand order repaired and checked by reading, not compiled, not run on hardware
//
#include <hip/hip_runtime.h>

// ---------------------------------------------------------------------------
// LayoutGraphModel on MI455X (gfx1250).
// bf16 activations/weights + v_wmma_f32_16x16x32_bf16 (f32 accum) for all 8
// GEMMs; fp32 global atomics for edge mean-aggregation; LDS reduction for the
// per-row L2 norm; bf16 streaming halves HBM traffic on the bandwidth-bound
// scatter/gather phases (23.3 TB/s roofline). GEMM kernel is templated on
// tiles-per-wave so the WMMA loop is branch-free with EXEC all-ones.
// ---------------------------------------------------------------------------

#define NN 131072          // nodes
#define EE 786432          // edges
#define SEGSZ 1024         // nodes per segment
#define NSEG_ 128
#define S_ 16              // segments per graph

typedef __attribute__((ext_vector_type(16))) __bf16 v16bf;
typedef __attribute__((ext_vector_type(8)))  float  v8f;

union BPack { uint4 q[2]; v16bf v; };

__device__ __forceinline__ float bf2f(unsigned short u) {
  unsigned int x = ((unsigned int)u) << 16;
  return __builtin_bit_cast(float, x);
}
__device__ __forceinline__ unsigned short f2bf(float f) {
  unsigned int x = __builtin_bit_cast(unsigned int, f);
  x += 0x7FFFu + ((x >> 16) & 1u);        // round-to-nearest-even
  return (unsigned short)(x >> 16);
}

// --- weight transpose + bf16 convert: W[K][Nout] -> Wt[NoutPad][Kpad] -------
__global__ void k_prep_w(const float* __restrict__ W, int K, int Nout,
                         int Kpad, int NoutPad, unsigned short* __restrict__ Wt) {
  int gid = blockIdx.x * blockDim.x + threadIdx.x;
  int total = NoutPad * Kpad;
  if (gid >= total) return;
  int n = gid / Kpad, k = gid - n * Kpad;
  float v = (k < K && n < Nout) ? W[(size_t)k * Nout + n] : 0.f;
  Wt[(size_t)n * Kpad + k] = f2bf(v);
}

// --- feature concat: [nf(112) | ncf(14) | emb[op](64) | pad(2)] -> bf16 -----
__global__ void k_build_x0(const float* __restrict__ nf, const float* __restrict__ ncf,
                           const int* __restrict__ ops, const float* __restrict__ emb,
                           unsigned short* __restrict__ X0) {
  int gid = blockIdx.x * blockDim.x + threadIdx.x;
  if (gid >= NN * 192) return;
  int n = gid / 192, f = gid - n * 192;
  float v;
  if (f < 112)      v = nf[(size_t)n * 112 + f];
  else if (f < 126) v = ncf[(size_t)n * 14 + (f - 112)];
  else if (f < 190) v = emb[(size_t)ops[n] * 64 + (f - 126)];
  else              v = 0.f;
  X0[gid] = f2bf(v);
}

// --- generic WMMA GEMM: Out = epi(A1@B1t [+ A2@B2t] + bias) -----------------
// A: bf16 [N][lda] row-major (K padded to x32). Bt: bf16 [Nout][ldb] (W^T).
// Block: 16 rows x Nout cols, 8 waves; wave w owns n-tiles {w, w+8, ...}.
// TPW = tiles per wave (Nout = TPW*128), compile-time so the WMMA loop is
// branch-free (EXEC all-ones, no accumulator merge movs).
template <int TPW, bool HAS2, bool RELU, bool L2>
__global__ __launch_bounds__(256)
void k_gemm(const unsigned short* __restrict__ A1, int lda1,
            const unsigned short* __restrict__ B1, int ldb1, int K1,
            const unsigned short* __restrict__ A2, int lda2,
            const unsigned short* __restrict__ B2, int ldb2, int K2,
            const float* __restrict__ bias, int nbias,
            unsigned short* __restrict__ Out, int ldo) {
  __shared__ float rowsq[16];
  const int tid  = threadIdx.x;
  const int wave = tid >> 5;
  const int lane = tid & 31;
  const int hlf  = lane >> 4;      // which 16-lane half
  const int ml   = lane & 15;      // A-row / B-col within tile
  const int m0   = blockIdx.x * 16;

  v8f acc[TPW] = {};

#pragma unroll
  for (int pass = 0; pass < (HAS2 ? 2 : 1); ++pass) {
    const unsigned short* A = pass ? A2 : A1;
    const unsigned short* B = pass ? B2 : B1;
    const int lda = pass ? lda2 : lda1;
    const int ldb = pass ? ldb2 : ldb1;
    const int K   = pass ? K2   : K1;
    const unsigned short* arow = A + (size_t)(m0 + ml) * lda;
    for (int k0 = 0; k0 < K; k0 += 32) {
      // A operand (ISA layout): lanes0-15 hold K [k0..k0+7]&[k0+16..23],
      // lanes16-31 hold [k0+8..15]&[k0+24..31]  -> two contiguous b128 loads.
      BPack a;
      a.q[0] = *(const uint4*)(arow + k0 + hlf * 8);
      a.q[1] = *(const uint4*)(arow + k0 + 16 + hlf * 8);
      __builtin_prefetch(arow + k0 + 32, 0, 1);   // speculative; OOB dropped
#pragma unroll
      for (int t = 0; t < TPW; ++t) {
        // B operand: lane's column n = nt*16+ml, K = k0+hlf*8..+7 & k0+16+hlf*8..+7
        const int nt = wave + (t << 3);
        const unsigned short* bp = B + (size_t)(nt * 16 + ml) * ldb + k0 + hlf * 8;
        BPack b;
        b.q[0] = *(const uint4*)(bp);
        b.q[1] = *(const uint4*)(bp + 16);
        acc[t] = __builtin_amdgcn_wmma_f32_16x16x32_bf16(
            false, a.v, false, b.v, (short)0, acc[t], false, false);
      }
    }
  }

  // Epilogue: bias, (l2 norm over full row via LDS), relu, bf16 store.
  float val[TPW][8];
  float ss[8];
#pragma unroll
  for (int r = 0; r < 8; ++r) ss[r] = 0.f;
#pragma unroll
  for (int t = 0; t < TPW; ++t) {
    int col = (wave + (t << 3)) * 16 + ml;
    float bv = (col < nbias) ? bias[col] : 0.f;
#pragma unroll
    for (int r = 0; r < 8; ++r) {
      float v = acc[t][r] + bv;
      val[t][r] = v;
      ss[r] += v * v;
    }
  }

  if (L2) {
    if (tid < 16) rowsq[tid] = 0.f;
    __syncthreads();
#pragma unroll
    for (int r = 0; r < 8; ++r) atomicAdd(&rowsq[(hlf << 3) + r], ss[r]);
    __syncthreads();
  }

#pragma unroll
  for (int t = 0; t < TPW; ++t) {
    int col = (wave + (t << 3)) * 16 + ml;
#pragma unroll
    for (int r = 0; r < 8; ++r) {
      int row = m0 + (hlf << 3) + r;   // D layout: VGPR r -> M = r + 8*half
      float v = val[t][r];
      if (L2) v = v / fmaxf(sqrtf(rowsq[(hlf << 3) + r]), 1e-12f);
      if (RELU) v = fmaxf(v, 0.f);
      Out[(size_t)row * ldo + col] = f2bf(v);
    }
  }
}

// --- in-degree counts -------------------------------------------------------
__global__ void k_count(const int* __restrict__ edges, float* __restrict__ cnt) {
  int e = blockIdx.x * blockDim.x + threadIdx.x;
  if (e >= EE) return;
  atomicAdd(&cnt[edges[EE + e]], 1.f);
}

// --- edge scatter: msum[dst] += h[src] (4 features per thread) --------------
__global__ void k_scatter(const int* __restrict__ edges,
                          const unsigned short* __restrict__ H, int ld,
                          float* __restrict__ msum, int width) {
  long long gid = blockIdx.x * (long long)blockDim.x + threadIdx.x;
  int perEdge = width >> 2;
  long long tot = (long long)EE * perEdge;
  if (gid >= tot) return;
  int e = (int)(gid / perEdge);
  int c = (int)(gid - (long long)e * perEdge) << 2;
  int src = edges[e], dst = edges[EE + e];
  uint2 u = *(const uint2*)(H + (size_t)src * ld + c);
  float* mp = msum + (size_t)dst * width + c;
  atomicAdd(mp + 0, bf2f((unsigned short)(u.x & 0xffffu)));
  atomicAdd(mp + 1, bf2f((unsigned short)(u.x >> 16)));
  atomicAdd(mp + 2, bf2f((unsigned short)(u.y & 0xffffu)));
  atomicAdd(mp + 3, bf2f((unsigned short)(u.y >> 16)));
}

// --- mean = msum / max(cnt,1), bf16 -----------------------------------------
__global__ void k_mean(const float* __restrict__ msum, const float* __restrict__ cnt,
                       unsigned short* __restrict__ mean, int width) {
  long long gid = blockIdx.x * (long long)blockDim.x + threadIdx.x;
  long long tot = (long long)NN * width;
  if (gid >= tot) return;
  int n = (int)(gid / width);
  mean[gid] = f2bf(msum[gid] / fmaxf(cnt[n], 1.f));
}

// --- segment sum (1024 nodes x 512) + Linear(512->1) ------------------------
__global__ __launch_bounds__(256)
void k_seg(const unsigned short* __restrict__ H, const float* __restrict__ Wf,
           const float* __restrict__ bf_, const int* __restrict__ batches,
           float* __restrict__ out) {
  __shared__ float red[256];
  int seg = blockIdx.x, tid = threadIdx.x;
  int f0 = tid * 2, f1 = f0 + 1;
  float a0 = 0.f, a1 = 0.f;
  const unsigned short* base = H + (size_t)seg * SEGSZ * 512;
  for (int i = 0; i < SEGSZ; ++i) {
    unsigned int u = *(const unsigned int*)(base + (size_t)i * 512 + f0);
    a0 += bf2f((unsigned short)(u & 0xffffu));
    a1 += bf2f((unsigned short)(u >> 16));
  }
  red[tid] = a0 * Wf[f0] + a1 * Wf[f1];
  __syncthreads();
  for (int s = 128; s > 0; s >>= 1) {
    if (tid < s) red[tid] += red[tid + s];
    __syncthreads();
  }
  if (tid == 0) out[batches[seg] * S_ + (seg & (S_ - 1))] = red[0] + bf_[0];
}

// ---------------------------------------------------------------------------
extern "C" void kernel_launch(void* const* d_in, const int* in_sizes, int n_in,
                              void* d_out, int out_size, void* d_ws, size_t ws_size,
                              hipStream_t stream) {
  (void)in_sizes; (void)n_in; (void)out_size; (void)ws_size;
  const float* nf    = (const float*)d_in[0];
  const float* ncf   = (const float*)d_in[1];
  const int*   ops   = (const int*)d_in[2];
  const int*   edges = (const int*)d_in[3];
  const int*   batches = (const int*)d_in[5];
  const float* emb = (const float*)d_in[6];
  const float* W1  = (const float*)d_in[7];  const float* b1  = (const float*)d_in[8];
  const float* W2  = (const float*)d_in[9];  const float* b2  = (const float*)d_in[10];
  const float* Wl1 = (const float*)d_in[11]; const float* bl1 = (const float*)d_in[12];
  const float* Wr1 = (const float*)d_in[13];
  const float* Wl2 = (const float*)d_in[14]; const float* bl2 = (const float*)d_in[15];
  const float* Wr2 = (const float*)d_in[16];
  const float* Wl3 = (const float*)d_in[17]; const float* bl3 = (const float*)d_in[18];
  const float* Wr3 = (const float*)d_in[19];
  const float* Wf  = (const float*)d_in[20]; const float* bfp = (const float*)d_in[21];

  char* ws = (char*)d_ws;
  constexpr size_t MB = 1ull << 20;
  constexpr size_t OFF_W1T  = 0 * MB;
  constexpr size_t OFF_W2T  = 1 * MB;
  constexpr size_t OFF_WL1T = 2 * MB;
  constexpr size_t OFF_WR1T = 3 * MB;
  constexpr size_t OFF_WL2T = 4 * MB;
  constexpr size_t OFF_WR2T = 5 * MB;
  constexpr size_t OFF_WL3T = 6 * MB;
  constexpr size_t OFF_WR3T = 7 * MB;
  constexpr size_t OFF_X0   = 8 * MB;
  constexpr size_t OFF_H1   = OFF_X0  + (size_t)NN * 192 * 2;
  constexpr size_t OFF_H    = OFF_H1  + (size_t)NN * 384 * 2;
  constexpr size_t OFF_OUT  = OFF_H   + (size_t)NN * 512 * 2;
  constexpr size_t OFF_MEAN = OFF_OUT + (size_t)NN * 512 * 2;
  constexpr size_t OFF_MSUM = OFF_MEAN + (size_t)NN * 512 * 2;
  constexpr size_t OFF_CNT  = OFF_MSUM + (size_t)NN * 512 * 4;

  unsigned short* W1T  = (unsigned short*)(ws + OFF_W1T);
  unsigned short* W2T  = (unsigned short*)(ws + OFF_W2T);
  unsigned short* WL1T = (unsigned short*)(ws + OFF_WL1T);
  unsigned short* WR1T = (unsigned short*)(ws + OFF_WR1T);
  unsigned short* WL2T = (unsigned short*)(ws + OFF_WL2T);
  unsigned short* WR2T = (unsigned short*)(ws + OFF_WR2T);
  unsigned short* WL3T = (unsigned short*)(ws + OFF_WL3T);
  unsigned short* WR3T = (unsigned short*)(ws + OFF_WR3T);
  unsigned short* X0   = (unsigned short*)(ws + OFF_X0);
  unsigned short* H1   = (unsigned short*)(ws + OFF_H1);
  unsigned short* H    = (unsigned short*)(ws + OFF_H);
  unsigned short* OUT  = (unsigned short*)(ws + OFF_OUT);
  unsigned short* MEAN = (unsigned short*)(ws + OFF_MEAN);
  float*          MSUM = (float*)(ws + OFF_MSUM);
  float*          CNT  = (float*)(ws + OFF_CNT);

  auto cdiv = [](long long a, long long b) { return (unsigned)((a + b - 1) / b); };

  // weights -> transposed bf16
  k_prep_w<<<cdiv(384 * 192, 256), 256, 0, stream>>>(W1, 190, 380, 192, 384, W1T);
  k_prep_w<<<cdiv(256 * 384, 256), 256, 0, stream>>>(W2, 380, 256, 384, 256, W2T);
  k_prep_w<<<cdiv(512 * 256, 256), 256, 0, stream>>>(Wl1, 256, 512, 256, 512, WL1T);
  k_prep_w<<<cdiv(512 * 256, 256), 256, 0, stream>>>(Wr1, 256, 512, 256, 512, WR1T);
  k_prep_w<<<cdiv(512 * 512, 256), 256, 0, stream>>>(Wl2, 512, 512, 512, 512, WL2T);
  k_prep_w<<<cdiv(512 * 512, 256), 256, 0, stream>>>(Wr2, 512, 512, 512, 512, WR2T);
  k_prep_w<<<cdiv(512 * 512, 256), 256, 0, stream>>>(Wl3, 512, 512, 512, 512, WL3T);
  k_prep_w<<<cdiv(512 * 512, 256), 256, 0, stream>>>(Wr3, 512, 512, 512, 512, WR3T);

  // feature assembly
  k_build_x0<<<cdiv((long long)NN * 192, 256), 256, 0, stream>>>(nf, ncf, ops, emb, X0);

  const unsigned NB = NN / 16;  // 8192 row-blocks
  // MLP: relu(x@W1+b1) then l2norm(h@W2+b2)
  k_gemm<3, false, true, false><<<NB, 256, 0, stream>>>(
      X0, 192, W1T, 192, 192, nullptr, 0, nullptr, 0, 0, b1, 380, H1, 384);
  k_gemm<2, false, false, true><<<NB, 256, 0, stream>>>(
      H1, 384, W2T, 384, 384, nullptr, 0, nullptr, 0, 0, b2, 256, H, 256);

  // in-degree counts (shared by all 3 SAGE layers)
  hipMemsetAsync(CNT, 0, (size_t)NN * 4, stream);
  k_count<<<cdiv(EE, 256), 256, 0, stream>>>(edges, CNT);

  // SAGE layer 1 (256 -> 512): relu(l2norm(mean@Wl1 + bl1 + h@Wr1))
  hipMemsetAsync(MSUM, 0, (size_t)NN * 256 * 4, stream);
  k_scatter<<<cdiv((long long)EE * 64, 256), 256, 0, stream>>>(edges, H, 256, MSUM, 256);
  k_mean<<<cdiv((long long)NN * 256, 256), 256, 0, stream>>>(MSUM, CNT, MEAN, 256);
  k_gemm<4, true, true, true><<<NB, 256, 0, stream>>>(
      MEAN, 256, WL1T, 256, 256, H, 256, WR1T, 256, 256, bl1, 512, OUT, 512);

  // SAGE layer 2 (512 -> 512)
  hipMemsetAsync(MSUM, 0, (size_t)NN * 512 * 4, stream);
  k_scatter<<<cdiv((long long)EE * 128, 256), 256, 0, stream>>>(edges, OUT, 512, MSUM, 512);
  k_mean<<<cdiv((long long)NN * 512, 256), 256, 0, stream>>>(MSUM, CNT, MEAN, 512);
  k_gemm<4, true, true, true><<<NB, 256, 0, stream>>>(
      MEAN, 512, WL2T, 512, 512, OUT, 512, WR2T, 512, 512, bl2, 512, H, 512);

  // SAGE layer 3 (512 -> 512), no relu
  hipMemsetAsync(MSUM, 0, (size_t)NN * 512 * 4, stream);
  k_scatter<<<cdiv((long long)EE * 128, 256), 256, 0, stream>>>(edges, H, 512, MSUM, 512);
  k_mean<<<cdiv((long long)NN * 512, 256), 256, 0, stream>>>(MSUM, CNT, MEAN, 512);
  k_gemm<4, true, false, true><<<NB, 256, 0, stream>>>(
      MEAN, 512, WL3T, 512, 512, H, 512, WR3T, 512, 512, bl3, 512, OUT, 512);

  // ragged segment sum + final linear
  k_seg<<<NSEG_, 256, 0, stream>>>(OUT, Wf, bfp, batches, (float*)d_out);
}
